// LlamaDecoderLayer_62801011802990
// MI455X (gfx1250) — matrix operand order repaired and checked by reading, not compiled, not run on hardware
//
#include <hip/hip_runtime.h>
#include <hip/hip_bf16.h>

typedef _Float16 half_t;
typedef __attribute__((ext_vector_type(16))) _Float16 v16h;
typedef __attribute__((ext_vector_type(8)))  _Float16 v8h;
typedef __attribute__((ext_vector_type(8)))  float    v8f;

#define BB 2
#define SS 2048
#define DD 2048
#define HH 16
#define HD 128
#define FF 8192
#define MROWS (BB * SS)          // 4096
#define LN_EPS 1e-5f
#define BIT_EPS 1e-5f

// ---------------------------------------------------------------------------
// Register-blocked wave-level WMMA GEMM:
//   C[z][M x N] = A[z][M x K] * B[z][N x K]^T
// Each wave32 owns a 64x64 macro-tile = 4x4 WMMA 16x16 tiles. Per K=32 step:
// 4 A-fragments + 4 B-fragments (16 x global_load_b128) feed 16 v_wmma ops,
// i.e. 1 b128 load per WMMA (4x less L2 traffic than unblocked).
// A is M-major (lda = K stride); B is stored N rows x K contiguous, matching
// the ISA 7.12.2 B-operand layout (lanes 0-15: K=k..k+7,k+16..k+23, lanes 16-31: +8).
// Per-z offsets are (z/hCount)*Outer + (z%hCount)*Inner so the context GEMM
// can scatter head outputs directly into the (B,S,D) layout.
// ---------------------------------------------------------------------------
template <bool OUTF16>
__global__ __launch_bounds__(256) void gemm_wmma_kernel(
    const half_t* __restrict__ A, const half_t* __restrict__ Bw,
    void* __restrict__ Cout, int M, int N, int K, int lda, int ldb, int ldc,
    long aOuter, long aInner, long bOuter, long bInner,
    long cOuter, long cInner, int hCount)
{
    const int lane = threadIdx.x & 31;
    const int wave = threadIdx.x >> 5;
    const long tile = (long)blockIdx.x * 8 + wave;   // 64x64 macro-tiles
    const int ntn = N >> 6;
    const int tM = (int)(tile / ntn);
    const int tN = (int)(tile % ntn);
    if (tM >= (M >> 6)) return;              // wave-uniform; EXEC stays all-ones

    const int z  = blockIdx.z;
    const int zo = z / hCount;
    const int zi = z % hCount;

    const half_t* Ab = A  + (long)zo * aOuter + (long)zi * aInner;
    const half_t* Bb = Bw + (long)zo * bOuter + (long)zi * bInner;

    // Fragment base pointers per ISA 7.12.2 VGPR layouts (wave32):
    //  A (16x32): lanes 0-15 -> K = k..k+7 and k+16..k+23; lanes 16-31 -> +8
    //  B (32x16): lanes 0-15 -> K = k..k+7 and k+16..k+23; lanes 16-31 -> +8
    const int mrow = (tM << 6) + (lane & 15);
    const int nrow = (tN << 6) + (lane & 15);
    const int hiA  = (lane >> 4) ? 8 : 0;
    const int hiB  = (lane >> 4) ? 8 : 0;
    const half_t* aptr = Ab + (long)mrow * lda + hiA;
    const half_t* bptr = Bb + (long)nrow * ldb + hiB;
    const long mstep = (long)16 * lda;       // row stride between sub-tiles
    const long nstep = (long)16 * ldb;

    v8f acc[4][4];
#pragma unroll
    for (int mi = 0; mi < 4; ++mi)
#pragma unroll
        for (int ni = 0; ni < 4; ++ni)
            acc[mi][ni] = v8f{0.f, 0.f, 0.f, 0.f, 0.f, 0.f, 0.f, 0.f};

    for (int k = 0; k < K; k += 32) {
        // prefetch next K panel into cache (global_prefetch_b8)
        __builtin_prefetch(aptr + k + 256, 0, 1);
        __builtin_prefetch(bptr + k + 256, 0, 1);

        v16h af[4], bf[4];
#pragma unroll
        for (int mi = 0; mi < 4; ++mi) {
            const half_t* p = aptr + mi * mstep + k;
            v8h a0 = *(const v8h*)(p);
            v8h a1 = *(const v8h*)(p + 16);
#pragma unroll
            for (int i = 0; i < 8; ++i) { af[mi][i] = a0[i]; af[mi][8 + i] = a1[i]; }
        }
#pragma unroll
        for (int ni = 0; ni < 4; ++ni) {
            const half_t* p = bptr + ni * nstep + k;
            v8h b0 = *(const v8h*)(p);
            v8h b1 = *(const v8h*)(p + 16);
#pragma unroll
            for (int i = 0; i < 8; ++i) { bf[ni][i] = b0[i]; bf[ni][8 + i] = b1[i]; }
        }
#pragma unroll
        for (int mi = 0; mi < 4; ++mi)
#pragma unroll
            for (int ni = 0; ni < 4; ++ni)
                acc[mi][ni] = __builtin_amdgcn_wmma_f32_16x16x32_f16(
                    false, af[mi], false, bf[ni], (short)0, acc[mi][ni],
                    false, false);
    }

    // C/D layout: VGPR j, lanes 0-15 -> M = j, lanes 16-31 -> M = j + 8
    const long cbase = (long)zo * cOuter + (long)zi * cInner;
#pragma unroll
    for (int mi = 0; mi < 4; ++mi) {
#pragma unroll
        for (int ni = 0; ni < 4; ++ni) {
            const int ccol  = (tN << 6) + (ni << 4) + (lane & 15);
            const int crow0 = (tM << 6) + (mi << 4) + ((lane >> 4) ? 8 : 0);
            if (OUTF16) {
                half_t* C = (half_t*)Cout;
#pragma unroll
                for (int j = 0; j < 8; ++j)
                    C[cbase + (long)(crow0 + j) * ldc + ccol] = (half_t)acc[mi][ni][j];
            } else {
                float* C = (float*)Cout;
#pragma unroll
                for (int j = 0; j < 8; ++j)
                    C[cbase + (long)(crow0 + j) * ldc + ccol] = acc[mi][ni][j];
            }
        }
    }
}

// ---------------------------------------------------------------------------
// Ternary weight quantization: scale = mean(|w|); tern = clip(round(w/scale))
// ---------------------------------------------------------------------------
__global__ void zero_kernel(float* p, int n)
{
    int i = blockIdx.x * blockDim.x + threadIdx.x;
    if (i < n) p[i] = 0.f;
}

__global__ __launch_bounds__(256) void absum_kernel(
    const float* __restrict__ w, long n, float* __restrict__ acc)
{
    __shared__ float red[256];
    int tid = threadIdx.x;
    float s = 0.f;
    for (long i = (long)blockIdx.x * 256 + tid; i < n; i += (long)gridDim.x * 256)
        s += fabsf(w[i]);
    red[tid] = s;
    __syncthreads();
    for (int st = 128; st > 0; st >>= 1) {
        if (tid < st) red[tid] += red[tid + st];
        __syncthreads();
    }
    if (tid == 0) atomicAdd(acc, red[0]);
}

__global__ __launch_bounds__(256) void quantize_kernel(
    const float* __restrict__ w, const float* __restrict__ acc, long n,
    half_t* __restrict__ out)
{
    long i = (long)blockIdx.x * 256 + threadIdx.x;
    if (i >= n) return;
    float scale = acc[0] / (float)n;
    float t = rintf(w[i] / (scale + BIT_EPS));
    t = fminf(1.f, fmaxf(-1.f, t));
    out[i] = (half_t)(t * scale);
}

// ---------------------------------------------------------------------------
// RMSNorm: one block per row of D=2048, f32 in -> f16 staged out
// ---------------------------------------------------------------------------
__global__ __launch_bounds__(256) void rmsnorm_kernel(
    const float* __restrict__ x, const float* __restrict__ w,
    half_t* __restrict__ out, int D)
{
    __shared__ float red[256];
    const long row = blockIdx.x;
    const int tid = threadIdx.x;
    const float* xr = x + row * D;
    float s = 0.f;
    for (int c = tid; c < D; c += 256) { float v = xr[c]; s += v * v; }
    red[tid] = s;
    __syncthreads();
    for (int st = 128; st > 0; st >>= 1) {
        if (tid < st) red[tid] += red[tid + st];
        __syncthreads();
    }
    float inv = rsqrtf(red[0] / (float)D + LN_EPS);
    half_t* orow = out + row * D;
    for (int c = tid; c < D; c += 256)
        orow[c] = (half_t)(xr[c] * inv * w[c]);
}

// ---------------------------------------------------------------------------
// RoPE: (B,S,D) f16 q,k -> (B,H,S,HD) f16 staged for attention GEMMs
// ---------------------------------------------------------------------------
__global__ __launch_bounds__(256) void rope_kernel(
    const half_t* __restrict__ Qf, const half_t* __restrict__ Kf,
    const float* __restrict__ cosT, const float* __restrict__ sinT,
    half_t* __restrict__ Qh, half_t* __restrict__ Kh)
{
    long idx = (long)blockIdx.x * 256 + threadIdx.x;      // over B*S*H*(HD/2)
    const long total = (long)BB * SS * HH * (HD / 2);
    if (idx >= total) return;
    int i = (int)(idx % (HD / 2));
    int h = (int)((idx / (HD / 2)) % HH);
    int s = (int)((idx / ((HD / 2) * (long)HH)) % SS);
    int b = (int)(idx / ((HD / 2) * (long)HH * SS));

    float c  = cosT[(long)s * HD + i];       // halves duplicated in table
    float sn = sinT[(long)s * HD + i];
    long ibase = ((long)b * SS + s) * DD + (long)h * HD;
    long obase = (((long)b * HH + h) * SS + s) * (long)HD;

    float q1 = (float)Qf[ibase + i], q2 = (float)Qf[ibase + i + HD / 2];
    Qh[obase + i]          = (half_t)(q1 * c - q2 * sn);
    Qh[obase + i + HD / 2] = (half_t)(q2 * c + q1 * sn);

    float k1 = (float)Kf[ibase + i], k2 = (float)Kf[ibase + i + HD / 2];
    Kh[obase + i]          = (half_t)(k1 * c - k2 * sn);
    Kh[obase + i + HD / 2] = (half_t)(k2 * c + k1 * sn);
}

// V: (B,S,D) -> Vt (B,H,HD,S) so the PV GEMM's B-operand K (=t) is contiguous
__global__ __launch_bounds__(256) void vtrans_kernel(
    const half_t* __restrict__ Vf, half_t* __restrict__ Vt)
{
    long idx = (long)blockIdx.x * 256 + threadIdx.x;      // over B*H*HD*S
    const long total = (long)BB * HH * HD * SS;
    if (idx >= total) return;
    int s  = (int)(idx % SS);
    int hd = (int)((idx / SS) % HD);
    int h  = (int)((idx / ((long)SS * HD)) % HH);
    int b  = (int)(idx / ((long)SS * HD * HH));
    Vt[idx] = Vf[((long)b * SS + s) * DD + (long)h * HD + hd];
}

// ---------------------------------------------------------------------------
// Causal softmax in place on f16 scores: one block per (z, row)
// ---------------------------------------------------------------------------
__global__ __launch_bounds__(256) void softmax_kernel(
    half_t* __restrict__ Scores, float alpha)
{
    __shared__ float red[256];
    const int row = blockIdx.x;
    const int z   = blockIdx.y;
    const int tid = threadIdx.x;
    half_t* P = Scores + (long)z * SS * SS + (long)row * SS;

    float m = -3.0e38f;
    for (int c = tid; c <= row; c += 256)
        m = fmaxf(m, (float)P[c] * alpha);
    red[tid] = m;
    __syncthreads();
    for (int st = 128; st > 0; st >>= 1) {
        if (tid < st) red[tid] = fmaxf(red[tid], red[tid + st]);
        __syncthreads();
    }
    m = red[0];
    __syncthreads();

    float sum = 0.f;
    for (int c = tid; c <= row; c += 256) {
        float e = __expf((float)P[c] * alpha - m);
        P[c] = (half_t)e;
        sum += e;
    }
    red[tid] = sum;
    __syncthreads();
    for (int st = 128; st > 0; st >>= 1) {
        if (tid < st) red[tid] += red[tid + st];
        __syncthreads();
    }
    float inv = 1.f / red[0];
    for (int c = tid; c <= row; c += 256)
        P[c] = (half_t)((float)P[c] * inv);
    for (int c = row + 1 + tid; c < SS; c += 256)
        P[c] = (half_t)0.f;
}

// ---------------------------------------------------------------------------
// Elementwise helpers
// ---------------------------------------------------------------------------
__global__ __launch_bounds__(256) void add_f32_kernel(
    const float* __restrict__ a, const float* __restrict__ b,
    float* __restrict__ out, long n)
{
    long i = (long)blockIdx.x * 256 + threadIdx.x;
    if (i < n) out[i] = a[i] + b[i];
}

__global__ __launch_bounds__(256) void mul_f16_kernel(
    half_t* __restrict__ g, const half_t* __restrict__ u, long n)
{
    long i = (long)blockIdx.x * 256 + threadIdx.x;
    if (i < n) g[i] = (half_t)((float)g[i] * (float)u[i]);
}

// ---------------------------------------------------------------------------
// Host-side orchestration
// ---------------------------------------------------------------------------
extern "C" void kernel_launch(void* const* d_in, const int* in_sizes, int n_in,
                              void* d_out, int out_size, void* d_ws, size_t ws_size,
                              hipStream_t stream)
{
    (void)in_sizes; (void)n_in; (void)out_size; (void)ws_size;
    const float* hidden = (const float*)d_in[0];
    // d_in[1] = attention_mask (causal -1e9) — implemented analytically in softmax
    const float* cosT   = (const float*)d_in[2];
    const float* sinT   = (const float*)d_in[3];
    const float* w_raw[7] = {(const float*)d_in[4], (const float*)d_in[5],
                             (const float*)d_in[6], (const float*)d_in[7],
                             (const float*)d_in[8], (const float*)d_in[9],
                             (const float*)d_in[10]};
    const long  w_n[7] = {(long)DD * DD, (long)DD * DD, (long)DD * DD, (long)DD * DD,
                          (long)FF * DD, (long)FF * DD, (long)DD * FF};
    const float* ln1 = (const float*)d_in[11];
    const float* ln2 = (const float*)d_in[12];
    float* out = (float*)d_out;

    // -------- workspace carving (256B aligned) --------
    char* base = (char*)d_ws;
    size_t off = 0;
    auto alloc = [&](size_t bytes) -> void* {
        void* p = base + off;
        off = (off + bytes + 255) & ~(size_t)255;
        return p;
    };
    float*  scaleAcc = (float*)alloc(16 * sizeof(float));
    half_t* Wh[7];
    for (int i = 0; i < 7; ++i) Wh[i] = (half_t*)alloc((size_t)w_n[i] * 2);
    half_t* Xh  = (half_t*)alloc((size_t)MROWS * DD * 2);
    half_t* Qf  = (half_t*)alloc((size_t)MROWS * DD * 2);
    half_t* Kf  = (half_t*)alloc((size_t)MROWS * DD * 2);
    half_t* Vf  = (half_t*)alloc((size_t)MROWS * DD * 2);
    half_t* Qh  = (half_t*)alloc((size_t)MROWS * DD * 2);
    half_t* Kh  = (half_t*)alloc((size_t)MROWS * DD * 2);
    half_t* Vt  = (half_t*)alloc((size_t)MROWS * DD * 2);
    half_t* Scr = (half_t*)alloc((size_t)BB * HH * SS * SS * 2);
    half_t* Ctx = (half_t*)alloc((size_t)MROWS * DD * 2);
    float*  Apr = (float*)alloc((size_t)MROWS * DD * 4);
    float*  Hr  = (float*)alloc((size_t)MROWS * DD * 4);
    half_t* Yh  = (half_t*)alloc((size_t)MROWS * DD * 2);
    half_t* Gh  = (half_t*)alloc((size_t)MROWS * FF * 2);
    half_t* Uh  = (half_t*)alloc((size_t)MROWS * FF * 2);
    float*  Dwn = (float*)alloc((size_t)MROWS * DD * 4);

    auto cdiv = [](long a, long b) { return (int)((a + b - 1) / b); };

    // -------- 1) ternarize weights to f16 --------
    zero_kernel<<<1, 64, 0, stream>>>(scaleAcc, 16);
    for (int i = 0; i < 7; ++i)
        absum_kernel<<<1024, 256, 0, stream>>>(w_raw[i], w_n[i], scaleAcc + i);
    for (int i = 0; i < 7; ++i)
        quantize_kernel<<<cdiv(w_n[i], 256), 256, 0, stream>>>(
            w_raw[i], scaleAcc + i, w_n[i], Wh[i]);

    auto gemm16 = [&](const half_t* A, const half_t* Bw, half_t* C,
                      int M, int N, int K, int lda, int ldb, int ldc,
                      long aO, long aI, long bO, long bI, long cO, long cI,
                      int hC, int zN) {
        dim3 g((M / 64) * (N / 64) / 8, 1, zN);
        gemm_wmma_kernel<true><<<g, 256, 0, stream>>>(
            A, Bw, (void*)C, M, N, K, lda, ldb, ldc, aO, aI, bO, bI, cO, cI, hC);
    };
    auto gemm32 = [&](const half_t* A, const half_t* Bw, float* C,
                      int M, int N, int K, int lda, int ldb, int ldc,
                      long aO, long aI, long bO, long bI, long cO, long cI,
                      int hC, int zN) {
        dim3 g((M / 64) * (N / 64) / 8, 1, zN);
        gemm_wmma_kernel<false><<<g, 256, 0, stream>>>(
            A, Bw, (void*)C, M, N, K, lda, ldb, ldc, aO, aI, bO, bI, cO, cI, hC);
    };

    // -------- 2) RMSNorm 1 --------
    rmsnorm_kernel<<<MROWS, 256, 0, stream>>>(hidden, ln1, Xh, DD);

    // -------- 3) Q,K,V projections (f16 out) --------
    gemm16(Xh, Wh[0], Qf, MROWS, DD, DD, DD, DD, DD, 0,0, 0,0, 0,0, 1, 1);
    gemm16(Xh, Wh[1], Kf, MROWS, DD, DD, DD, DD, DD, 0,0, 0,0, 0,0, 1, 1);
    gemm16(Xh, Wh[2], Vf, MROWS, DD, DD, DD, DD, DD, 0,0, 0,0, 0,0, 1, 1);

    // -------- 4) RoPE + head staging --------
    rope_kernel<<<cdiv((long)BB * SS * HH * (HD / 2), 256), 256, 0, stream>>>(
        Qf, Kf, cosT, sinT, Qh, Kh);
    vtrans_kernel<<<cdiv((long)BB * HH * HD * SS, 256), 256, 0, stream>>>(Vf, Vt);

    // -------- 5) scores = Q K^T per (b,h), z = b*H + h --------
    gemm16(Qh, Kh, Scr, SS, SS, HD, HD, HD, SS,
           (long)SS * HD, 0, (long)SS * HD, 0, (long)SS * SS, 0, 1, BB * HH);

    // -------- 6) causal softmax (scale folded in) --------
    softmax_kernel<<<dim3(SS, BB * HH), 256, 0, stream>>>(
        Scr, 0.08838834764831845f /* 1/sqrt(128) */);

    // -------- 7) ctx = P V, scattered into (B,S,D) f16 --------
    gemm16(Scr, Vt, Ctx, SS, HD, SS, SS, SS, DD,
           (long)HH * SS * SS, (long)SS * SS,
           (long)HH * HD * SS, (long)HD * SS,
           (long)SS * DD, HD, HH, BB * HH);

    // -------- 8) output projection (f32) + residual --------
    gemm32(Ctx, Wh[3], Apr, MROWS, DD, DD, DD, DD, DD, 0,0, 0,0, 0,0, 1, 1);
    add_f32_kernel<<<cdiv((long)MROWS * DD, 256), 256, 0, stream>>>(
        hidden, Apr, Hr, (long)MROWS * DD);

    // -------- 9) RMSNorm 2 + gated FFN --------
    rmsnorm_kernel<<<MROWS, 256, 0, stream>>>(Hr, ln2, Yh, DD);
    gemm16(Yh, Wh[4], Gh, MROWS, FF, DD, DD, DD, FF, 0,0, 0,0, 0,0, 1, 1);
    gemm16(Yh, Wh[5], Uh, MROWS, FF, DD, DD, DD, FF, 0,0, 0,0, 0,0, 1, 1);
    mul_f16_kernel<<<cdiv((long)MROWS * FF, 256), 256, 0, stream>>>(
        Gh, Uh, (long)MROWS * FF);

    // -------- 10) down projection (f32) + final residual --------
    gemm32(Gh, Wh[6], Dwn, MROWS, DD, FF, FF, FF, DD, 0,0, 0,0, 0,0, 1, 1);
    add_f32_kernel<<<cdiv((long)MROWS * DD, 256), 256, 0, stream>>>(
        Hr, Dwn, out, (long)MROWS * DD);
}
